// CRISPR_RAG_Head_4827543241091
// MI455X (gfx1250) — matrix operand order repaired and checked by reading, not compiled, not run on hardware
//
#include <hip/hip_runtime.h>
#include <math.h>

// ---- problem constants (match reference) ----
#define B_Q    1024
#define DMODEL 768
#define MEMN   50000
#define KNEI   50
#define HEADS  8
#define HDIM   96
#define GHID   256
#define CHUNK  2048       // M-chunk for distance GEMM
#define BBLK   64         // queries per attention block

typedef __attribute__((ext_vector_type(8)))  __bf16 v8bf;
typedef __attribute__((ext_vector_type(16))) __bf16 v16bf;
typedef __attribute__((ext_vector_type(8)))  float  v8f;

__device__ __forceinline__ __bf16 f2bf(float x) {
    union { unsigned short u; __bf16 b; } c;
    unsigned int t = __float_as_uint(x);
    t += 0x7FFFu + ((t >> 16) & 1u);      // round-to-nearest-even
    c.u = (unsigned short)(t >> 16);
    return c.b;
}

// A-operand fragment: two 16B halves (K contiguous {16h+0..15})
__device__ __forceinline__ v16bf ld_a(const __bf16* __restrict__ p) {
    v8bf lo = *(const v8bf*)p;
    v8bf hi = *(const v8bf*)(p + 8);
    return __builtin_shufflevector(lo, hi, 0, 1, 2, 3, 4, 5, 6, 7,
                                   8, 9, 10, 11, 12, 13, 14, 15);
}

// 2x2 macro-tile (32x32 output) NT GEMM: 4 WMMAs per 8 b128 loads.
// a0/a1 = this lane's A-row pointers for M-tiles 0/1 (row = lane&15 within tile),
// b0/b1 = this lane's B-row pointers for N-tiles 0/1 (row = lane&15 == out col).
__device__ __forceinline__ void wmma_nt_2x2(const __bf16* __restrict__ a0,
                                            const __bf16* __restrict__ a1,
                                            const __bf16* __restrict__ b0,
                                            const __bf16* __restrict__ b1,
                                            int Kd, v8f& c00, v8f& c01,
                                            v8f& c10, v8f& c11) {
    const int half = (threadIdx.x >> 4) & 1;
    const __bf16* ap0 = a0 + half * 16;
    const __bf16* ap1 = a1 + half * 16;
    const __bf16* bp0 = b0 + half * 16;
    const __bf16* bp1 = b1 + half * 16;
    for (int k0 = 0; k0 < Kd; k0 += 32) {
        v16bf A0 = ld_a(ap0 + k0);
        v16bf A1 = ld_a(ap1 + k0);
        v16bf B0 = *(const v16bf*)(bp0 + k0);
        v16bf B1 = *(const v16bf*)(bp1 + k0);
        c00 = __builtin_amdgcn_wmma_f32_16x16x32_bf16(false, A0, false, B0, (short)0, c00, false, false);
        c01 = __builtin_amdgcn_wmma_f32_16x16x32_bf16(false, A0, false, B1, (short)0, c01, false, false);
        c10 = __builtin_amdgcn_wmma_f32_16x16x32_bf16(false, A1, false, B0, (short)0, c10, false, false);
        c11 = __builtin_amdgcn_wmma_f32_16x16x32_bf16(false, A1, false, B1, (short)0, c11, false, false);
    }
}

// 2x1 tail (32x16 output) for the odd 16-col remainder of the last chunk
__device__ __forceinline__ void wmma_nt_2x1(const __bf16* __restrict__ a0,
                                            const __bf16* __restrict__ a1,
                                            const __bf16* __restrict__ b0,
                                            int Kd, v8f& c00, v8f& c10) {
    const int half = (threadIdx.x >> 4) & 1;
    const __bf16* ap0 = a0 + half * 16;
    const __bf16* ap1 = a1 + half * 16;
    const __bf16* bp0 = b0 + half * 16;
    for (int k0 = 0; k0 < Kd; k0 += 32) {
        v16bf A0 = ld_a(ap0 + k0);
        v16bf A1 = ld_a(ap1 + k0);
        v16bf B0 = *(const v16bf*)(bp0 + k0);
        c00 = __builtin_amdgcn_wmma_f32_16x16x32_bf16(false, A0, false, B0, (short)0, c00, false, false);
        c10 = __builtin_amdgcn_wmma_f32_16x16x32_bf16(false, A1, false, B0, (short)0, c10, false, false);
    }
}

// -------------------- conversion / small kernels --------------------

__global__ void cvt_bf16_kernel(const float* __restrict__ src, __bf16* __restrict__ dst, int n) {
    int i = blockIdx.x * blockDim.x + threadIdx.x;
    if (i < n) dst[i] = f2bf(src[i]);
}

// gather neighbor-key rows through topk_idx and convert to bf16 in one pass
__global__ void gathercvt_kernel(const float* __restrict__ memk, const int* __restrict__ tkI,
                                 int bBase, __bf16* __restrict__ dst) {
    int i = blockIdx.x * blockDim.x + threadIdx.x;   // over BBLK*KNEI*DMODEL
    if (i >= BBLK * KNEI * DMODEL) return;
    int rl = i / DMODEL, d = i - rl * DMODEL;
    int b = bBase + rl / KNEI, k = rl % KNEI;
    dst[i] = f2bf(memk[(size_t)tkI[(size_t)b * KNEI + k] * DMODEL + d]);
}

__global__ void sqnorm_kernel(const float* __restrict__ X, float* __restrict__ out, int rows) {
    int r = blockIdx.x * blockDim.x + threadIdx.x;
    if (r >= rows) return;
    const float* x = X + (size_t)r * DMODEL;
    float s = 0.f;
    for (int d = 0; d < DMODEL; ++d) s += x[d] * x[d];
    out[r] = s;
}

__global__ void init_topk_kernel(float* __restrict__ tkD, int* __restrict__ tkI) {
    int i = blockIdx.x * blockDim.x + threadIdx.x;
    if (i < B_Q * KNEI) { tkD[i] = 3.4e38f; tkI[i] = 0; }
}

// -------------------- distance GEMM (WMMA, 32x32 macro tiles) --------------------
__global__ void dist_kernel(const __bf16* __restrict__ qbf, const __bf16* __restrict__ kbf,
                            const float* __restrict__ q2, const float* __restrict__ m2,
                            float* __restrict__ chunkD, int chunkBase, int cols) {
    int lane = threadIdx.x & 31;
    int wv   = threadIdx.x >> 5;
    int tnT  = cols >> 4;                 // 16-col tiles
    int tnM  = (tnT + 1) >> 1;            // 32-col macro tiles
    int tile = blockIdx.x * 4 + wv;
    if (tile >= (B_Q / 32) * tnM) return; // wave-uniform guard
    int tm = tile / tnM, tn = tile % tnM;
    int half = lane >> 4, l15 = lane & 15;
    int rBase = tm * 32, nBase = tn * 32;

    const __bf16* a0 = qbf + (size_t)(rBase + l15) * DMODEL;
    const __bf16* a1 = a0 + (size_t)16 * DMODEL;
    const __bf16* b0 = kbf + (size_t)(nBase + l15) * DMODEL;
    bool hasN1 = (2 * tn + 1) < tnT;      // wave-uniform

    v8f c00 = {}, c01 = {}, c10 = {}, c11 = {};
    if (hasN1) {
        wmma_nt_2x2(a0, a1, b0, b0 + (size_t)16 * DMODEL, DMODEL, c00, c01, c10, c11);
    } else {
        wmma_nt_2x1(a0, a1, b0, DMODEL, c00, c10);
    }

    int col0 = nBase + l15;
    float m20 = m2[chunkBase + col0];
    #pragma unroll
    for (int j = 0; j < 8; ++j) {
        int r = rBase + half * 8 + j;
        float d2a = q2[r] + m20 - 2.f * c00[j];
        chunkD[(size_t)r * CHUNK + col0] = __fsqrt_rn(fmaxf(d2a, 0.f));
        float d2b = q2[r + 16] + m20 - 2.f * c10[j];
        chunkD[(size_t)(r + 16) * CHUNK + col0] = __fsqrt_rn(fmaxf(d2b, 0.f));
    }
    if (hasN1) {
        int col1 = col0 + 16;
        float m21 = m2[chunkBase + col1];
        #pragma unroll
        for (int j = 0; j < 8; ++j) {
            int r = rBase + half * 8 + j;
            float d2a = q2[r] + m21 - 2.f * c01[j];
            chunkD[(size_t)r * CHUNK + col1] = __fsqrt_rn(fmaxf(d2a, 0.f));
            float d2b = q2[r + 16] + m21 - 2.f * c11[j];
            chunkD[(size_t)(r + 16) * CHUNK + col1] = __fsqrt_rn(fmaxf(d2b, 0.f));
        }
    }
}

// serial running top-K merge: one thread per query
__global__ void topk_merge_kernel(const float* __restrict__ chunkD, int chunkBase, int cols,
                                  float* __restrict__ tkD, int* __restrict__ tkI) {
    int b = blockIdx.x * blockDim.x + threadIdx.x;
    if (b >= B_Q) return;
    float* D = tkD + (size_t)b * KNEI;
    int*   I = tkI + (size_t)b * KNEI;
    float mx = -1.f; int mp = 0;
    for (int i = 0; i < KNEI; ++i) if (D[i] > mx) { mx = D[i]; mp = i; }
    const float* row = chunkD + (size_t)b * CHUNK;
    for (int c = 0; c < cols; ++c) {
        float d = row[c];
        if (d < mx) {
            D[mp] = d; I[mp] = chunkBase + c;
            mx = -1.f;
            for (int i = 0; i < KNEI; ++i) if (D[i] > mx) { mx = D[i]; mp = i; }
        }
    }
}

// -------------------- generic NT projection GEMM (WMMA, 32x32 macro tiles) ----
// C(rows x ncols) = A(rows x Kd) @ W(ncols x Kd)^T + bias; rows%32==0, ncols%32==0
__global__ void gemm_nt_bias_kernel(const __bf16* __restrict__ A, const __bf16* __restrict__ W,
                                    const float* __restrict__ bias, float* __restrict__ C,
                                    int rows, int Kd, int ncols) {
    int lane = threadIdx.x & 31;
    int wv   = threadIdx.x >> 5;
    int tnM  = ncols >> 5;
    int tile = blockIdx.x * 4 + wv;
    if (tile >= (rows >> 5) * tnM) return;
    int tm = tile / tnM, tn = tile % tnM;
    int half = lane >> 4, l15 = lane & 15;
    int rBase = tm * 32, nBase = tn * 32;

    const __bf16* a0 = A + (size_t)(rBase + l15) * Kd;
    const __bf16* b0 = W + (size_t)(nBase + l15) * Kd;
    v8f c00 = {}, c01 = {}, c10 = {}, c11 = {};
    wmma_nt_2x2(a0, a0 + (size_t)16 * Kd, b0, b0 + (size_t)16 * Kd, Kd, c00, c01, c10, c11);

    int col0 = nBase + l15, col1 = col0 + 16;
    float bv0 = bias[col0], bv1 = bias[col1];
    #pragma unroll
    for (int j = 0; j < 8; ++j) {
        int r = rBase + half * 8 + j;
        C[(size_t)r * ncols + col0]        = c00[j] + bv0;
        C[(size_t)r * ncols + col1]        = c01[j] + bv1;
        C[(size_t)(r + 16) * ncols + col0] = c10[j] + bv0;
        C[(size_t)(r + 16) * ncols + col1] = c11[j] + bv1;
    }
}

// -------------------- per-(b,h) wave attention --------------------
__global__ void attn_kernel(const float* __restrict__ qp, const float* __restrict__ kpb,
                            const float* __restrict__ vpb, float* __restrict__ ctxw, int bBase) {
    __shared__ float sw[8][64];
    int lane = threadIdx.x & 31;
    int wv   = threadIdx.x >> 5;
    int gw   = blockIdx.x * 8 + wv;               // 0 .. BBLK*HEADS-1 (exact)
    int bl   = gw >> 3;
    int h    = gw & 7;
    int b    = bBase + bl;
    const float* q = qp + (size_t)b * DMODEL + h * HDIM;

    for (int k = lane; k < KNEI; k += 32) {
        const float* kr = kpb + (size_t)(bl * KNEI + k) * DMODEL + h * HDIM;
        float s = 0.f;
        for (int d = 0; d < HDIM; ++d) s += q[d] * kr[d];
        sw[wv][k] = s * 0.1020620726159658f;      // 1/sqrt(96)
    }
    __syncthreads();
    float mx = -3.4e38f;
    for (int k = 0; k < KNEI; ++k) mx = fmaxf(mx, sw[wv][k]);
    float sum = 0.f;
    for (int k = 0; k < KNEI; ++k) sum += __expf(sw[wv][k] - mx);
    __syncthreads();
    for (int k = lane; k < KNEI; k += 32) sw[wv][k] = __expf(sw[wv][k] - mx) / sum;
    __syncthreads();

    float a0 = 0.f, a1 = 0.f, a2 = 0.f;
    int dbase = lane * 3;                         // 96 = 32 * 3
    for (int k = 0; k < KNEI; ++k) {
        float w = sw[wv][k];
        const float* vr = vpb + (size_t)(bl * KNEI + k) * DMODEL + h * HDIM + dbase;
        a0 += w * vr[0]; a1 += w * vr[1]; a2 += w * vr[2];
    }
    float* o = ctxw + (size_t)b * DMODEL + h * HDIM + dbase;
    o[0] = a0; o[1] = a1; o[2] = a2;
}

// -------------------- output projection + residual (WMMA) --------------------
__global__ void outproj_kernel(const __bf16* __restrict__ ctxwbf, const __bf16* __restrict__ Wobf,
                               const float* __restrict__ bo, const float* __restrict__ query,
                               float* __restrict__ context, float* __restrict__ outEnh) {
    int lane = threadIdx.x & 31;
    int wv   = threadIdx.x >> 5;
    const int tnM = DMODEL >> 5;                  // 24
    int tile = blockIdx.x * 4 + wv;
    if (tile >= (B_Q >> 5) * tnM) return;
    int tm = tile / tnM, tn = tile % tnM;
    int half = lane >> 4, l15 = lane & 15;
    int rBase = tm * 32, nBase = tn * 32;

    const __bf16* a0 = ctxwbf + (size_t)(rBase + l15) * DMODEL;
    const __bf16* b0 = Wobf + (size_t)(nBase + l15) * DMODEL;
    v8f c00 = {}, c01 = {}, c10 = {}, c11 = {};
    wmma_nt_2x2(a0, a0 + (size_t)16 * DMODEL, b0, b0 + (size_t)16 * DMODEL,
                DMODEL, c00, c01, c10, c11);

    int col0 = nBase + l15, col1 = col0 + 16;
    float bv0 = bo[col0], bv1 = bo[col1];
    #pragma unroll
    for (int j = 0; j < 8; ++j) {
        int r0 = rBase + half * 8 + j;
        int r1 = r0 + 16;
        float v00 = c00[j] + bv0, v01 = c01[j] + bv1;
        float v10 = c10[j] + bv0, v11 = c11[j] + bv1;
        context[(size_t)r0 * DMODEL + col0] = v00;
        context[(size_t)r0 * DMODEL + col1] = v01;
        context[(size_t)r1 * DMODEL + col0] = v10;
        context[(size_t)r1 * DMODEL + col1] = v11;
        outEnh[(size_t)r0 * DMODEL + col0] = query[(size_t)r0 * DMODEL + col0] + v00;
        outEnh[(size_t)r0 * DMODEL + col1] = query[(size_t)r0 * DMODEL + col1] + v01;
        outEnh[(size_t)r1 * DMODEL + col0] = query[(size_t)r1 * DMODEL + col0] + v10;
        outEnh[(size_t)r1 * DMODEL + col1] = query[(size_t)r1 * DMODEL + col1] + v11;
    }
}

// -------------------- gate MLP layer 1 (WMMA, piecewise K) --------------------
__global__ void gate1_kernel(const __bf16* __restrict__ qbf, const __bf16* __restrict__ ctxbf,
                             const __bf16* __restrict__ Wg1bf, const float* __restrict__ bg1,
                             float* __restrict__ gh) {
    int lane = threadIdx.x & 31;
    int wv   = threadIdx.x >> 5;
    const int tnM = GHID >> 5;                    // 8
    int tile = blockIdx.x * 4 + wv;
    if (tile >= (B_Q >> 5) * tnM) return;
    int tm = tile / tnM, tn = tile % tnM;
    int half = lane >> 4, l15 = lane & 15;
    int rBase = tm * 32, nBase = tn * 32;

    const __bf16* aq = qbf   + (size_t)(rBase + l15) * DMODEL;
    const __bf16* ac = ctxbf + (size_t)(rBase + l15) * DMODEL;
    const __bf16* b0 = Wg1bf + (size_t)(nBase + l15) * (2 * DMODEL);
    const __bf16* b1 = b0 + (size_t)16 * (2 * DMODEL);

    v8f c00 = {}, c01 = {}, c10 = {}, c11 = {};
    wmma_nt_2x2(aq, aq + (size_t)16 * DMODEL, b0, b1, DMODEL, c00, c01, c10, c11);
    wmma_nt_2x2(ac, ac + (size_t)16 * DMODEL, b0 + DMODEL, b1 + DMODEL,
                DMODEL, c00, c01, c10, c11);

    int col0 = nBase + l15, col1 = col0 + 16;
    float bv0 = bg1[col0], bv1 = bg1[col1];
    #pragma unroll
    for (int j = 0; j < 8; ++j) {
        int r = rBase + half * 8 + j;
        gh[(size_t)r * GHID + col0]        = fmaxf(c00[j] + bv0, 0.f);
        gh[(size_t)r * GHID + col1]        = fmaxf(c01[j] + bv1, 0.f);
        gh[(size_t)(r + 16) * GHID + col0] = fmaxf(c10[j] + bv0, 0.f);
        gh[(size_t)(r + 16) * GHID + col1] = fmaxf(c11[j] + bv1, 0.f);
    }
}

// -------------------- distance mean + kNN regression + gate2 --------------------
__global__ void sum_partial_kernel(const float* __restrict__ tkD, float* __restrict__ part) {
    __shared__ float sh[256];
    int i = blockIdx.x * 256 + threadIdx.x;       // B*K = 51200 = 200 blocks exactly
    sh[threadIdx.x] = tkD[i];
    __syncthreads();
    for (int s = 128; s > 0; s >>= 1) {
        if (threadIdx.x < s) sh[threadIdx.x] += sh[threadIdx.x + s];
        __syncthreads();
    }
    if (threadIdx.x == 0) part[blockIdx.x] = sh[0];
}

__global__ void finalize_mean_kernel(const float* __restrict__ part, float* __restrict__ meanOut) {
    if (threadIdx.x == 0) {
        float s = 0.f;
        for (int i = 0; i < 200; ++i) s += part[i];
        meanOut[0] = s / (float)(B_Q * KNEI);
    }
}

__global__ void rag_kernel(const float* __restrict__ tkD, const int* __restrict__ tkI,
                           const float* __restrict__ memv, const float* __restrict__ meanPtr,
                           float* __restrict__ outPred) {
    int b = blockIdx.x * blockDim.x + threadIdx.x;
    if (b >= B_Q) return;
    float inv = 1.f / meanPtr[0];
    const float* D = tkD + (size_t)b * KNEI;
    const int*   I = tkI + (size_t)b * KNEI;
    float ws = 0.f, acc = 0.f;
    for (int k = 0; k < KNEI; ++k) {
        float w = __expf(-D[k] * inv);
        ws += w;
        acc += w * memv[I[k]];
    }
    outPred[b] = acc / ws;
}

__global__ void gate2_kernel(const float* __restrict__ gh, const float* __restrict__ Wg2,
                             const float* __restrict__ bg2, float* __restrict__ outGate) {
    int b = blockIdx.x * blockDim.x + threadIdx.x;
    if (b >= B_Q) return;
    const float* g = gh + (size_t)b * GHID;
    float s = bg2[0];
    for (int i = 0; i < GHID; ++i) s += g[i] * Wg2[i];
    outGate[b] = 1.f / (1.f + __expf(-s));
}

// -------------------- host orchestration --------------------
extern "C" void kernel_launch(void* const* d_in, const int* in_sizes, int n_in,
                              void* d_out, int out_size, void* d_ws, size_t ws_size,
                              hipStream_t stream) {
    (void)in_sizes; (void)n_in; (void)out_size; (void)ws_size;

    const float* query = (const float*)d_in[0];
    const float* memk  = (const float*)d_in[1];
    const float* memv  = (const float*)d_in[2];
    const float* Wq  = (const float*)d_in[3];  const float* bq  = (const float*)d_in[4];
    const float* Wk  = (const float*)d_in[5];  const float* bk  = (const float*)d_in[6];
    const float* Wv  = (const float*)d_in[7];  const float* bv  = (const float*)d_in[8];
    const float* Wo  = (const float*)d_in[9];  const float* bo  = (const float*)d_in[10];
    const float* Wg1 = (const float*)d_in[11]; const float* bg1 = (const float*)d_in[12];
    const float* Wg2 = (const float*)d_in[13]; const float* bg2 = (const float*)d_in[14];

    float* outEnh  = (float*)d_out;
    float* outPred = outEnh + (size_t)B_Q * DMODEL;
    float* outGate = outPred + B_Q;

    char* ws = (char*)d_ws;
    size_t off = 0;
    auto take = [&](size_t bytes) -> char* {
        char* p = ws + off;
        off = (off + bytes + 255) & ~(size_t)255;
        return p;
    };
    // fp32 buffers
    float* m2    = (float*)take((size_t)MEMN * 4);
    float* q2    = (float*)take((size_t)B_Q * 4);
    float* tkD   = (float*)take((size_t)B_Q * KNEI * 4);
    int*   tkI   = (int*)  take((size_t)B_Q * KNEI * 4);
    float* qp    = (float*)take((size_t)B_Q * DMODEL * 4);
    float* ctxw  = (float*)take((size_t)B_Q * DMODEL * 4);
    float* ctx   = (float*)take((size_t)B_Q * DMODEL * 4);
    float* gh    = (float*)take((size_t)B_Q * GHID * 4);
    float* part  = (float*)take(256 * 4);
    float* meanv = (float*)take(256);
    // bf16 operand buffers (converted once, loaded many times)
    __bf16* qbf    = (__bf16*)take((size_t)B_Q * DMODEL * 2);
    __bf16* Wqbf   = (__bf16*)take((size_t)DMODEL * DMODEL * 2);
    __bf16* Wkbf   = (__bf16*)take((size_t)DMODEL * DMODEL * 2);
    __bf16* Wvbf   = (__bf16*)take((size_t)DMODEL * DMODEL * 2);
    __bf16* Wobf   = (__bf16*)take((size_t)DMODEL * DMODEL * 2);
    __bf16* Wg1bf  = (__bf16*)take((size_t)GHID * 2 * DMODEL * 2);
    __bf16* kbf    = (__bf16*)take((size_t)CHUNK * DMODEL * 2);       // per-chunk keys
    __bf16* nkbf   = (__bf16*)take((size_t)BBLK * KNEI * DMODEL * 2); // gathered neighbors
    __bf16* ctxwbf = (__bf16*)take((size_t)B_Q * DMODEL * 2);
    __bf16* ctxbf  = (__bf16*)take((size_t)B_Q * DMODEL * 2);
    // phase-overlaid scratch: chunkD (8 MB) then kpb/vpb (19.7 MB)
    char* scratch = take((size_t)2 * BBLK * KNEI * DMODEL * 4);
    float* chunkD = (float*)scratch;
    float* kpb    = (float*)scratch;
    float* vpb    = kpb + (size_t)BBLK * KNEI * DMODEL;

    const int T = 256;
    auto nb = [](int n, int t) { return (n + t - 1) / t; };

    // ---- squared norms + init + one-time bf16 conversions ----
    sqnorm_kernel<<<nb(MEMN, T), T, 0, stream>>>(memk, m2, MEMN);
    sqnorm_kernel<<<nb(B_Q, T), T, 0, stream>>>(query, q2, B_Q);
    init_topk_kernel<<<nb(B_Q * KNEI, T), T, 0, stream>>>(tkD, tkI);
    cvt_bf16_kernel<<<nb(B_Q * DMODEL, T), T, 0, stream>>>(query, qbf, B_Q * DMODEL);
    cvt_bf16_kernel<<<nb(DMODEL * DMODEL, T), T, 0, stream>>>(Wq, Wqbf, DMODEL * DMODEL);
    cvt_bf16_kernel<<<nb(DMODEL * DMODEL, T), T, 0, stream>>>(Wk, Wkbf, DMODEL * DMODEL);
    cvt_bf16_kernel<<<nb(DMODEL * DMODEL, T), T, 0, stream>>>(Wv, Wvbf, DMODEL * DMODEL);
    cvt_bf16_kernel<<<nb(DMODEL * DMODEL, T), T, 0, stream>>>(Wo, Wobf, DMODEL * DMODEL);
    cvt_bf16_kernel<<<nb(GHID * 2 * DMODEL, T), T, 0, stream>>>(Wg1, Wg1bf, GHID * 2 * DMODEL);

    // ---- chunked distance GEMM + top-K merge ----
    for (int cb = 0; cb < MEMN; cb += CHUNK) {
        int cols = (MEMN - cb < CHUNK) ? (MEMN - cb) : CHUNK;    // 2048 or 848, mult of 16
        cvt_bf16_kernel<<<nb(cols * DMODEL, T), T, 0, stream>>>(
            memk + (size_t)cb * DMODEL, kbf, cols * DMODEL);
        int tnM   = ((cols >> 4) + 1) >> 1;                      // 32-col macro tiles
        int tiles = (B_Q / 32) * tnM;
        dist_kernel<<<nb(tiles, 4), 128, 0, stream>>>(qbf, kbf, q2, m2, chunkD, cb, cols);
        topk_merge_kernel<<<nb(B_Q, T), T, 0, stream>>>(chunkD, cb, cols, tkD, tkI);
    }

    // ---- q projection ----
    {
        int tiles = (B_Q / 32) * (DMODEL / 32);
        gemm_nt_bias_kernel<<<nb(tiles, 4), 128, 0, stream>>>(qbf, Wqbf, bq, qp,
                                                              B_Q, DMODEL, DMODEL);
    }

    // ---- attention, blocked over queries ----
    for (int bBase = 0; bBase < B_Q; bBase += BBLK) {
        gathercvt_kernel<<<nb(BBLK * KNEI * DMODEL, T), T, 0, stream>>>(memk, tkI, bBase, nkbf);
        int tiles = ((BBLK * KNEI) / 32) * (DMODEL / 32);        // 2400
        gemm_nt_bias_kernel<<<nb(tiles, 4), 128, 0, stream>>>(nkbf, Wkbf, bk, kpb,
                                                              BBLK * KNEI, DMODEL, DMODEL);
        gemm_nt_bias_kernel<<<nb(tiles, 4), 128, 0, stream>>>(nkbf, Wvbf, bv, vpb,
                                                              BBLK * KNEI, DMODEL, DMODEL);
        attn_kernel<<<(BBLK * HEADS) / 8, 256, 0, stream>>>(qp, kpb, vpb, ctxw, bBase);
    }

    // ---- output projection + residual into d_out ----
    cvt_bf16_kernel<<<nb(B_Q * DMODEL, T), T, 0, stream>>>(ctxw, ctxwbf, B_Q * DMODEL);
    {
        int tiles = (B_Q / 32) * (DMODEL / 32);
        outproj_kernel<<<nb(tiles, 4), 128, 0, stream>>>(ctxwbf, Wobf, bo, query, ctx, outEnh);
    }

    // ---- kNN regression (global distance mean, then per-query weights) ----
    sum_partial_kernel<<<(B_Q * KNEI) / 256, 256, 0, stream>>>(tkD, part);
    finalize_mean_kernel<<<1, 32, 0, stream>>>(part, meanv);
    rag_kernel<<<nb(B_Q, T), T, 0, stream>>>(tkD, tkI, memv, meanv, outPred);

    // ---- gate MLP ----
    cvt_bf16_kernel<<<nb(B_Q * DMODEL, T), T, 0, stream>>>(ctx, ctxbf, B_Q * DMODEL);
    {
        int tiles = (B_Q / 32) * (GHID / 32);
        gate1_kernel<<<nb(tiles, 4), 128, 0, stream>>>(qbf, ctxbf, Wg1bf, bg1, gh);
    }
    gate2_kernel<<<nb(B_Q, T), T, 0, stream>>>(gh, Wg2, bg2, outGate);
}
